// DependencyGraphAnalyzer_59133109731856
// MI455X (gfx1250) — matrix operand order repaired and checked by reading, not compiled, not run on hardware
//
#include <hip/hip_runtime.h>
#include <hip/hip_bf16.h>
#include <stdint.h>

#define N_NODES 20000
#define N_EDGES 320000
#define IN_DIM  256
#define HID     128
#define HEADS   4

typedef __attribute__((ext_vector_type(16))) __bf16 v16bf;
typedef __attribute__((ext_vector_type(8)))  float  v8f;

// ---------- helpers ----------
static __device__ __forceinline__ __bf16 f2bf(float f) {
  unsigned u = __float_as_uint(f);
  unsigned r = u + 0x7FFFu + ((u >> 16) & 1u);   // round-to-nearest-even
  unsigned short s = (unsigned short)(r >> 16);
  return __builtin_bit_cast(__bf16, s);
}
static __device__ __forceinline__ unsigned f2ord(float f) {
  unsigned u = __float_as_uint(f);
  return (u & 0x80000000u) ? ~u : (u | 0x80000000u);
}
static __device__ __forceinline__ float ord2f(unsigned o) {
  unsigned u = (o & 0x80000000u) ? (o ^ 0x80000000u) : ~o;
  return __uint_as_float(u);
}

// ---------- elementwise ----------
__global__ void k_f32_to_bf16(const float* __restrict__ in, __bf16* __restrict__ out, int n) {
  int i = blockIdx.x * blockDim.x + threadIdx.x;
  int stride = gridDim.x * blockDim.x;
  for (; i < n; i += stride) out[i] = f2bf(in[i]);
}

// W[K,N] f32 -> Bt[N,K] bf16 (weights are tiny; done once per launch)
__global__ void k_w_transpose_bf16(const float* __restrict__ W, __bf16* __restrict__ Bt,
                                   int K, int N) {
  int i = blockIdx.x * blockDim.x + threadIdx.x;
  if (i >= K * N) return;
  int n = i / K, k = i - n * K;
  Bt[i] = f2bf(W[(size_t)k * N + n]);
}

__global__ void k_zero(unsigned* __restrict__ p, int n) {
  int i = blockIdx.x * blockDim.x + threadIdx.x;
  int stride = gridDim.x * blockDim.x;
  for (; i < n; i += stride) p[i] = 0u;
}

__global__ void k_bias_elu(float* __restrict__ h, const float* __restrict__ bias, int n, int D) {
  int i = blockIdx.x * blockDim.x + threadIdx.x;
  if (i >= n) return;
  float v = h[i] + bias[i % D];
  h[i] = (v > 0.f) ? v : expm1f(v);
}

__global__ void k_bias_only(float* __restrict__ h, const float* __restrict__ bias, int n, int D) {
  int i = blockIdx.x * blockDim.x + threadIdx.x;
  if (i >= n) return;
  h[i] = h[i] + bias[i % D];
}

// ---------- WMMA GEMM: C[M,N] = A[M,K](bf16) @ Bt[N,K]^T(bf16) + bias[N], fp32 out ----
// block = 128 threads = 4 waves; block tile 128(M) x 64(N); K step 32.
// A: LDS double-buffered (fragment layout needs cross-thread shuffle).
// B: fragments loaded DIRECTLY from global per lane — Bt is tiny and WGP$/L2-resident;
//    this removes the rb prefetch registers (which the allocator kept spilling to scratch)
//    and the whole lB LDS buffer.
__global__ __launch_bounds__(128, 1)
void k_wmma_gemm_bias(const __bf16* __restrict__ A, const __bf16* __restrict__ Bt,
                      const float* __restrict__ bias, float* __restrict__ C,
                      int M, int N, int K) {
  __shared__ __bf16 lA[2][128][32];

  const int t = threadIdx.x;
  const int wave = t >> 5;
  const int lane = t & 31;
  const int m0 = blockIdx.x * 128;
  const int n0 = blockIdx.y * 64;

  v8f acc[2][4];
  #pragma unroll
  for (int i = 0; i < 2; ++i)
    #pragma unroll
    for (int j = 0; j < 4; ++j) acc[i][j] = (v8f){0,0,0,0,0,0,0,0};

  const int mr  = lane & 15;
  const int kb  = (lane < 16) ? 0 : 8;    // A frag: K runs {0-7,16-23} / {8-15,24-31}
  const int kb2 = (lane < 16) ? 0 : 8;    // B frag: K runs {0-7,16-23} / {8-15,24-31}

  // per-lane B fragment row pointers (loop-invariant; k0 added per step)
  const __bf16* pb[4];
  #pragma unroll
  for (int j = 0; j < 4; ++j)
    pb[j] = Bt + (size_t)(n0 + (j << 4) + mr) * K + kb2;

  uint4 ra[4];
  auto load_A = [&](int k0) {
    uint4 z; z.x = z.y = z.z = z.w = 0u;
    ra[0] = ra[1] = ra[2] = ra[3] = z;
    int gr = m0 + t;                                   // one full 32-K row per thread
    if (gr < M) {
      const uint4* pa = (const uint4*)(A + (size_t)gr * K + k0);
      ra[0] = pa[0]; ra[1] = pa[1]; ra[2] = pa[2]; ra[3] = pa[3];
    }
  };
  auto store_A = [&](int p) {
    *(uint4*)&lA[p][t][0]  = ra[0];
    *(uint4*)&lA[p][t][8]  = ra[1];
    *(uint4*)&lA[p][t][16] = ra[2];
    *(uint4*)&lA[p][t][24] = ra[3];
  };

  load_A(0);
  store_A(0);
  __syncthreads();

  int p = 0;
  for (int k0 = 0; k0 < K; k0 += 32) {
    const bool more = (k0 + 32 < K);
    if (more) load_A(k0 + 32);             // prefetch next A tile into registers

    // ---- A fragments from LDS buffer p ----
    v16bf af0, af1;
    ((uint4*)&af0)[0] = *(const uint4*)&lA[p][(wave << 5) + mr][kb];
    ((uint4*)&af0)[1] = *(const uint4*)&lA[p][(wave << 5) + mr][kb + 16];
    ((uint4*)&af1)[0] = *(const uint4*)&lA[p][(wave << 5) + 16 + mr][kb];
    ((uint4*)&af1)[1] = *(const uint4*)&lA[p][(wave << 5) + 16 + mr][kb + 16];

    // ---- B fragments straight from global (cache-resident weights) ----
    v16bf bf[4];
    #pragma unroll
    for (int j = 0; j < 4; ++j) {
      ((uint4*)&bf[j])[0] = *(const uint4*)(pb[j] + k0);
      ((uint4*)&bf[j])[1] = *(const uint4*)(pb[j] + k0 + 16);
    }

    // ---- retire A prefetch regs into the other buffer (ends ra liveness) ----
    if (more) store_A(p ^ 1);

    // ---- 8 back-to-back WMMAs ----
    #pragma unroll
    for (int j = 0; j < 4; ++j) {
      acc[0][j] = __builtin_amdgcn_wmma_f32_16x16x32_bf16(
          false, af0, false, bf[j], (short)0, acc[0][j], false, false);
      acc[1][j] = __builtin_amdgcn_wmma_f32_16x16x32_bf16(
          false, af1, false, bf[j], (short)0, acc[1][j], false, false);
    }

    __syncthreads();
    p ^= 1;
  }

  // ---- store: D layout — VGPR r: row = base + r + 8*(lane>=16), col = lane&15 ----
  const int nc = lane & 15;
  #pragma unroll
  for (int i = 0; i < 2; ++i) {
    const int rbase = m0 + (wave << 5) + (i << 4) + ((lane >> 4) << 3);
    #pragma unroll
    for (int j = 0; j < 4; ++j) {
      int col = n0 + (j << 4) + nc;
      float bv = bias[col];
      #pragma unroll
      for (int r = 0; r < 8; ++r) {
        int row = rbase + r;
        if (row < M) C[(size_t)row * N + col] = acc[i][j][r] + bv;
      }
    }
  }
}

// ---------- GATv2 edge scoring: score[e,h] = att_h . leaky_relu(xl[src]+xr[dst]) ----------
template <int H, int C>
__global__ __launch_bounds__(256)
void k_edge_score(const float* __restrict__ xl, const float* __restrict__ xr,
                  const long long* __restrict__ src, const long long* __restrict__ dst,
                  const float* __restrict__ att, float* __restrict__ score, int E) {
  constexpr int D = H * C;
  constexpr int PER = D / 32;
  constexpr int G = 32 / H;      // lanes per head
  const int lane = threadIdx.x & 31;
  const int e = blockIdx.x * 8 + (threadIdx.x >> 5);
  if (e >= E) return;
  const long long is = src[e], id = dst[e];
  const float4* pl = (const float4*)(xl + (size_t)is * D + lane * PER);
  const float4* pr = (const float4*)(xr + (size_t)id * D + lane * PER);
  const float4* pa = (const float4*)(att + lane * PER);  // att flat [H*C] == channel idx
  float s = 0.f;
  #pragma unroll
  for (int q = 0; q < PER / 4; ++q) {
    float4 a = pl[q], b = pr[q], w = pa[q];
    float v;
    v = a.x + b.x; v = (v > 0.f) ? v : 0.2f * v; s += w.x * v;
    v = a.y + b.y; v = (v > 0.f) ? v : 0.2f * v; s += w.y * v;
    v = a.z + b.z; v = (v > 0.f) ? v : 0.2f * v; s += w.z * v;
    v = a.w + b.w; v = (v > 0.f) ? v : 0.2f * v; s += w.w * v;
  }
  #pragma unroll
  for (int off = G >> 1; off > 0; off >>= 1) s += __shfl_xor(s, off, 32);
  if ((lane & (G - 1)) == 0) score[(size_t)e * H + (lane / G)] = s;
}

template <int H>
__global__ __launch_bounds__(256)
void k_seg_max(const float* __restrict__ score, const long long* __restrict__ dst,
               unsigned* __restrict__ smax, int E) {
  int i = blockIdx.x * blockDim.x + threadIdx.x;
  if (i >= E * H) return;
  int e = i / H, h = i - e * H;
  atomicMax(&smax[(size_t)dst[e] * H + h], f2ord(score[i]));
}

template <int H>
__global__ __launch_bounds__(256)
void k_exp_denom(float* __restrict__ score, const long long* __restrict__ dst,
                 const unsigned* __restrict__ smax, float* __restrict__ denom, int E) {
  int i = blockIdx.x * blockDim.x + threadIdx.x;
  if (i >= E * H) return;
  int e = i / H, h = i - e * H;
  float m = ord2f(smax[(size_t)dst[e] * H + h]);
  float ex = __expf(score[i] - m);
  score[i] = ex;
  atomicAdd(&denom[(size_t)dst[e] * H + h], ex);
}

template <int H, int C>
__global__ __launch_bounds__(256)
void k_aggregate(const float* __restrict__ xl, const long long* __restrict__ src,
                 const long long* __restrict__ dst, const float* __restrict__ ex,
                 const float* __restrict__ denom, float* __restrict__ out, int E) {
  constexpr int D = H * C;
  constexpr int PER = D / 32;
  const int lane = threadIdx.x & 31;
  const int e = blockIdx.x * 8 + (threadIdx.x >> 5);
  if (e >= E) return;
  const long long is = src[e], id = dst[e];
  const int h0 = (lane * PER) / C;
  float alpha = ex[(size_t)e * H + h0] / (denom[(size_t)id * H + h0] + 1e-16f);
  const float4* pl = (const float4*)(xl + (size_t)is * D + lane * PER);
  float* po = out + (size_t)id * D + lane * PER;
  #pragma unroll
  for (int q = 0; q < PER / 4; ++q) {
    float4 a = pl[q];
    atomicAdd(&po[q * 4 + 0], alpha * a.x);
    atomicAdd(&po[q * 4 + 1], alpha * a.y);
    atomicAdd(&po[q * 4 + 2], alpha * a.z);
    atomicAdd(&po[q * 4 + 3], alpha * a.w);
  }
}

// ---------- output heads: sigmoid(h@Wa+ba), sigmoid(h@Wrc+brc) ----------
__global__ __launch_bounds__(256)
void k_heads(const float* __restrict__ h, const float* __restrict__ Wa, const float* __restrict__ ba,
             const float* __restrict__ Wrc, const float* __restrict__ brc,
             float* __restrict__ out, int N) {
  const int lane = threadIdx.x & 31;
  const int node = blockIdx.x * 8 + (threadIdx.x >> 5);
  if (node >= N) return;
  const float4* ph = (const float4*)(h + (size_t)node * HID + lane * 4);
  const float4* wa = (const float4*)(Wa + lane * 4);
  const float4* wr = (const float4*)(Wrc + lane * 4);
  float4 v = ph[0], a = wa[0], r = wr[0];
  float s1 = v.x * a.x + v.y * a.y + v.z * a.z + v.w * a.w;
  float s2 = v.x * r.x + v.y * r.y + v.z * r.z + v.w * r.w;
  #pragma unroll
  for (int off = 16; off > 0; off >>= 1) {
    s1 += __shfl_xor(s1, off, 32);
    s2 += __shfl_xor(s2, off, 32);
  }
  if (lane == 0) {
    out[node]     = 1.f / (1.f + __expf(-(s1 + ba[0])));
    out[N + node] = 1.f / (1.f + __expf(-(s2 + brc[0])));
  }
}

// ---------- host orchestration ----------
static inline int cdiv(int a, int b) { return (a + b - 1) / b; }

extern "C" void kernel_launch(void* const* d_in, const int* in_sizes, int n_in,
                              void* d_out, int out_size, void* d_ws, size_t ws_size,
                              hipStream_t stream) {
  const float*     x    = (const float*)d_in[0];
  const long long* ei   = (const long long*)d_in[1];
  const long long* src  = ei;
  const long long* dst  = ei + N_EDGES;
  const float* W1l  = (const float*)d_in[2];  const float* b1l   = (const float*)d_in[3];
  const float* W1r  = (const float*)d_in[4];  const float* b1r   = (const float*)d_in[5];
  const float* att1 = (const float*)d_in[6];  const float* bias1 = (const float*)d_in[7];
  const float* W2l  = (const float*)d_in[8];  const float* b2l   = (const float*)d_in[9];
  const float* W2r  = (const float*)d_in[10]; const float* b2r   = (const float*)d_in[11];
  const float* att2 = (const float*)d_in[12]; const float* bias2 = (const float*)d_in[13];
  const float* Wa   = (const float*)d_in[14]; const float* ba    = (const float*)d_in[15];
  const float* Wrc  = (const float*)d_in[16]; const float* brc   = (const float*)d_in[17];
  float* out = (float*)d_out;

  const int D1 = HEADS * HID;  // 512

  // workspace carve-up (stream-ordered aliasing)
  char* ws = (char*)d_ws;
  size_t off = 0;
  auto take = [&](size_t bytes) -> char* {
    char* p = ws + off;
    off += (bytes + 255) & ~(size_t)255;
    return p;
  };
  __bf16* xbf   = (__bf16*)take((size_t)N_NODES * D1 * 2);   // x_bf16, later h_bf16
  __bf16* w1lbf = (__bf16*)take((size_t)IN_DIM * D1 * 2);    // transposed [N][K]
  __bf16* w1rbf = (__bf16*)take((size_t)IN_DIM * D1 * 2);
  __bf16* w2lbf = (__bf16*)take((size_t)D1 * HID * 2);
  __bf16* w2rbf = (__bf16*)take((size_t)D1 * HID * 2);
  float* R2 = (float*)take((size_t)N_NODES * D1 * 4);        // xl1; later xl2|xr2|out2
  float* R3 = (float*)take((size_t)N_NODES * D1 * 4);        // xr1; later out1 (= h)
  float* score   = (float*)take((size_t)N_EDGES * HEADS * 4);
  unsigned* smax = (unsigned*)take((size_t)N_NODES * HEADS * 4);
  float* denom   = (float*)take((size_t)N_NODES * HEADS * 4);

  float* xl1  = R2;
  float* xr1  = R3;
  float* out1 = R3;                       // reuse xr1 region after scoring
  float* xl2  = R2;
  float* xr2  = R2 + (size_t)N_NODES * HID;
  float* out2 = R2 + (size_t)2 * N_NODES * HID;

  const int EB = cdiv(N_EDGES, 8);        // wave-per-edge blocks (256 thr = 8 waves)

  // ===== layer 1 =====
  k_f32_to_bf16<<<cdiv(N_NODES * IN_DIM, 256), 256, 0, stream>>>(x, xbf, N_NODES * IN_DIM);
  k_w_transpose_bf16<<<cdiv(IN_DIM * D1, 256), 256, 0, stream>>>(W1l, w1lbf, IN_DIM, D1);
  k_w_transpose_bf16<<<cdiv(IN_DIM * D1, 256), 256, 0, stream>>>(W1r, w1rbf, IN_DIM, D1);
  k_w_transpose_bf16<<<cdiv(D1 * HID, 256), 256, 0, stream>>>(W2l, w2lbf, D1, HID);
  k_w_transpose_bf16<<<cdiv(D1 * HID, 256), 256, 0, stream>>>(W2r, w2rbf, D1, HID);

  {
    dim3 g(cdiv(N_NODES, 128), D1 / 64);
    k_wmma_gemm_bias<<<g, 128, 0, stream>>>(xbf, w1lbf, b1l, xl1, N_NODES, D1, IN_DIM);
    k_wmma_gemm_bias<<<g, 128, 0, stream>>>(xbf, w1rbf, b1r, xr1, N_NODES, D1, IN_DIM);
  }

  k_zero<<<cdiv(N_NODES * HEADS, 256), 256, 0, stream>>>(smax, N_NODES * HEADS);
  k_zero<<<cdiv(N_NODES * HEADS, 256), 256, 0, stream>>>((unsigned*)denom, N_NODES * HEADS);
  k_edge_score<HEADS, HID><<<EB, 256, 0, stream>>>(xl1, xr1, src, dst, att1, score, N_EDGES);
  k_zero<<<cdiv(N_NODES * D1, 256), 256, 0, stream>>>((unsigned*)out1, N_NODES * D1); // xr1 dead
  k_seg_max<HEADS><<<cdiv(N_EDGES * HEADS, 256), 256, 0, stream>>>(score, dst, smax, N_EDGES);
  k_exp_denom<HEADS><<<cdiv(N_EDGES * HEADS, 256), 256, 0, stream>>>(score, dst, smax, denom, N_EDGES);
  k_aggregate<HEADS, HID><<<EB, 256, 0, stream>>>(xl1, src, dst, score, denom, out1, N_EDGES);
  k_bias_elu<<<cdiv(N_NODES * D1, 256), 256, 0, stream>>>(out1, bias1, N_NODES * D1, D1);

  // ===== layer 2 =====
  k_f32_to_bf16<<<cdiv(N_NODES * D1, 256), 256, 0, stream>>>(out1, xbf, N_NODES * D1); // h_bf16
  {
    dim3 g(cdiv(N_NODES, 128), HID / 64);
    k_wmma_gemm_bias<<<g, 128, 0, stream>>>(xbf, w2lbf, b2l, xl2, N_NODES, HID, D1);
    k_wmma_gemm_bias<<<g, 128, 0, stream>>>(xbf, w2rbf, b2r, xr2, N_NODES, HID, D1);
  }

  k_zero<<<cdiv(N_NODES, 256), 256, 0, stream>>>(smax, N_NODES);
  k_zero<<<cdiv(N_NODES, 256), 256, 0, stream>>>((unsigned*)denom, N_NODES);
  k_zero<<<cdiv(N_NODES * HID, 256), 256, 0, stream>>>((unsigned*)out2, N_NODES * HID);
  k_edge_score<1, HID><<<EB, 256, 0, stream>>>(xl2, xr2, src, dst, att2, score, N_EDGES);
  k_seg_max<1><<<cdiv(N_EDGES, 256), 256, 0, stream>>>(score, dst, smax, N_EDGES);
  k_exp_denom<1><<<cdiv(N_EDGES, 256), 256, 0, stream>>>(score, dst, smax, denom, N_EDGES);
  k_aggregate<1, HID><<<EB, 256, 0, stream>>>(xl2, src, dst, score, denom, out2, N_EDGES);
  k_bias_only<<<cdiv(N_NODES * HID, 256), 256, 0, stream>>>(out2, bias2, N_NODES * HID, HID);

  // ===== heads =====
  k_heads<<<cdiv(N_NODES, 8), 256, 0, stream>>>(out2, Wa, ba, Wrc, brc, out, N_NODES);
}
